// PositionalInferenceBlock_28578712388214
// MI455X (gfx1250) — matrix operand order repaired and checked by reading, not compiled, not run on hardware
//
#include <hip/hip_runtime.h>
#include <hip/hip_bf16.h>
#include <cstddef>

// ---------------------------------------------------------------------------
// CDNA5 (gfx1250) wave32 WMMA helpers: D(16x16,f32) = A(16x32,bf16)@B(32x16,bf16)+C
// ---------------------------------------------------------------------------
typedef __attribute__((ext_vector_type(16))) __bf16 v16bf;
typedef __attribute__((ext_vector_type(8)))  float  v8f;
typedef __attribute__((ext_vector_type(4)))  int    v4i;

__device__ __forceinline__ v8f wmma_bf16(v16bf a, v16bf b, v8f c) {
  // (neg_a, A, neg_b, B, c_mod, C, reuse_a, reuse_b)
  return __builtin_amdgcn_wmma_f32_16x16x32_bf16(false, a, false, b, (short)0, c,
                                                 false, false);
}

// A fragment: 16(M) x 32(K), source row-major bf16 s[m*ldm + k].
// Per lane: two contiguous 8-element runs -> 2x ds_load_b128.
__device__ __forceinline__ v16bf ld_frag_a(const __bf16* s, int ldm, int lane) {
  v16bf a;
  const int m  = lane & 15;
  const int hk = (lane < 16) ? 0 : 8;
#pragma unroll
  for (int e = 0; e < 16; ++e) {
    const int k = (e < 8) ? (hk + e) : (16 + hk + (e - 8));
    a[e] = s[m * ldm + k];
  }
  return a;
}

// B fragment from TRANSPOSED storage: s[n*ldk + k] (== B[k][n]).
// Per lane: two contiguous 8-element runs -> 2x ds_load_b128.
__device__ __forceinline__ v16bf ld_frag_bT(const __bf16* s, int ldk, int lane) {
  v16bf b;
  const int n  = lane & 15;
  const int kb = (lane < 16) ? 0 : 8;
#pragma unroll
  for (int e = 0; e < 16; ++e) b[e] = s[n * ldk + ((e < 8) ? (kb + e) : (16 + kb + (e - 8)))];
  return b;
}

__device__ __forceinline__ float wave_sum(float v) {
#pragma unroll
  for (int o = 16; o > 0; o >>= 1) v += __shfl_xor(v, o, 32);
  return v;
}
__device__ __forceinline__ float wave_max(float v) {
#pragma unroll
  for (int o = 16; o > 0; o >>= 1) v = fmaxf(v, __shfl_xor(v, o, 32));
  return v;
}

// ---------------------------------------------------------------------------
// CDNA5 async global->LDS copy path (ASYNCcnt), guarded so the build cannot
// break on a toolchain without the builtin. Probe round 2 showed this builtin
// takes generic `v4i*` pointers on this toolchain.
// ---------------------------------------------------------------------------
#if defined(__HIP_DEVICE_COMPILE__) && \
    __has_builtin(__builtin_amdgcn_global_load_async_to_lds_b128)
#define USE_ASYNC_COPY 1
#else
#define USE_ASYNC_COPY 0
#endif

__device__ __forceinline__ void wait_async0() {
#if defined(__HIP_DEVICE_COMPILE__) && __has_builtin(__builtin_amdgcn_s_wait_asynccnt)
  __builtin_amdgcn_s_wait_asynccnt(0);
#elif defined(__HIP_DEVICE_COMPILE__)
  asm volatile("s_wait_asynccnt 0x0" ::: "memory");
#endif
}

// ---------------------------------------------------------------------------
// Kernel 1: per-row precompute over all 4096 (t,s) rows.
//   tpb  = [X@Wa+ba | X@Wb0+bb]   (f32, 4096x64; theta cols 0..31, PhiB 32..63)
//   G    = X@Wg0                  (bf16, row-major g_bf AND per-time-block
//                                  transposed g_tr[i][f][k] for async B-staging)
//   Qt   = theta@Wb1^T ; R9 = Wp@Qt, c = Qt.bp ; Rt10 = Wt@Qt, ct = Qt.bt
// ---------------------------------------------------------------------------
__global__ __launch_bounds__(256) void k_precompute(
    const float* __restrict__ batch, const float* __restrict__ Wa,
    const float* __restrict__ ba, const float* __restrict__ Wb,
    const float* __restrict__ bb, const float* __restrict__ Wg,
    const float* __restrict__ Wp, const float* __restrict__ bp,
    const float* __restrict__ Wt, const float* __restrict__ bt,
    float* __restrict__ tpb_all, __bf16* __restrict__ g_bf,
    __bf16* __restrict__ g_tr, float* __restrict__ r9c,
    float* __restrict__ rt10c) {
  __shared__ __align__(16) __bf16 sX[64 * 128];   // 16 KB  input rows (A, row-major)
  __shared__ __align__(16) __bf16 sWT[128 * 128]; // 32 KB  weights, TRANSPOSED [n][k]
  __shared__ float sTh[64 * 32];                  //  8 KB  theta tile (derived stage)
  __shared__ float sWb1[32 * 32];                 //  4 KB  Wb rows 128..159
  __shared__ float sBias[64];                     //  256 B [ba | bb]

  const int tid = threadIdx.x, wave = tid >> 5, lane = tid & 31;
  const int r0 = blockIdx.x * 64;

  for (int idx = tid; idx < 64 * 128; idx += 256)
    sX[idx] = (__bf16)batch[(size_t)r0 * 128 + idx];
  // phase A weights transposed: sWT[n*128+k] = [Wa|Wb0][k][n], n<64
  for (int idx = tid; idx < 64 * 128; idx += 256) {
    const int n = idx >> 7, k = idx & 127;
    sWT[idx] = (__bf16)((n < 32) ? Wa[k * 32 + n] : Wb[k * 32 + (n - 32)]);
  }
  for (int idx = tid; idx < 32 * 32; idx += 256)
    sWb1[idx] = Wb[(128 + (idx >> 5)) * 32 + (idx & 31)];
  if (tid < 64) sBias[tid] = (tid < 32) ? ba[tid] : bb[tid - 32];
  __syncthreads();

  // Phase A: (64x128)@(128x64) -> [theta | PhiB]. 16 tiles, K = 4 WMMA steps.
  for (int t = wave; t < 16; t += 8) {
    const int mt = t >> 2, nt = t & 3;
    v8f acc = {};
#pragma unroll
    for (int kk = 0; kk < 4; ++kk) {
      v16bf a = ld_frag_a(sX + mt * 16 * 128 + kk * 32, 128, lane);
      v16bf b = ld_frag_bT(sWT + nt * 16 * 128 + kk * 32, 128, lane);
      acc = wmma_bf16(a, b, acc);
    }
    const int n = nt * 16 + (lane & 15);
    const float bias = sBias[n];
#pragma unroll
    for (int v = 0; v < 8; ++v) {
      const int m = mt * 16 + v + ((lane < 16) ? 0 : 8);
      const float val = acc[v] + bias;
      tpb_all[(size_t)(r0 + m) * 64 + n] = val;
      if (n < 32) sTh[m * 32 + n] = val;
    }
  }
  __syncthreads();

  // Phase B: stage Wg0 transposed (sWT[n*128+k] = Wg[k][n]), compute G = X@Wg0.
  for (int idx = tid; idx < 128 * 128; idx += 256) {
    const int n = idx >> 7, k = idx & 127;
    sWT[idx] = (__bf16)Wg[k * 128 + n];
  }
  __syncthreads();
  for (int t = wave; t < 32; t += 8) {
    const int mt = t >> 3, nt = t & 7;
    v8f acc = {};
#pragma unroll
    for (int kk = 0; kk < 4; ++kk) {
      v16bf a = ld_frag_a(sX + mt * 16 * 128 + kk * 32, 128, lane);
      v16bf b = ld_frag_bT(sWT + nt * 16 * 128 + kk * 32, 128, lane);
      acc = wmma_bf16(a, b, acc);
    }
    const int n = nt * 16 + (lane & 15);
#pragma unroll
    for (int v = 0; v < 8; ++v) {
      const int m = mt * 16 + v + ((lane < 16) ? 0 : 8);
      const __bf16 g = (__bf16)acc[v];
      g_bf[(size_t)(r0 + m) * 128 + n] = g;                       // [t*s][f]
      g_tr[(size_t)blockIdx.x * 8192 + n * 64 + m] = g;           // [i][f][k]
    }
  }
  __syncthreads();

  // Derived tiny per-row vectors (one row per thread).
  if (tid < 64) {
    const int r = r0 + tid;
    float q[32];
#pragma unroll
    for (int b = 0; b < 32; ++b) {
      float a = 0.f;
#pragma unroll
      for (int k = 0; k < 32; ++k) a += sTh[tid * 32 + k] * sWb1[b * 32 + k];
      q[b] = a;
    }
    float c = 0.f, ct = 0.f;
#pragma unroll
    for (int b = 0; b < 32; ++b) { c += q[b] * bp[b]; ct += q[b] * bt[b]; }
#pragma unroll
    for (int d = 0; d < 9; ++d) {
      float a = 0.f;
#pragma unroll
      for (int b = 0; b < 32; ++b) a += Wp[d * 32 + b] * q[b];
      r9c[(size_t)r * 10 + d] = a;
    }
    r9c[(size_t)r * 10 + 9] = c;
#pragma unroll
    for (int d = 0; d < 10; ++d) {
      float a = 0.f;
#pragma unroll
      for (int b = 0; b < 32; ++b) a += Wt[d * 32 + b] * q[b];
      rt10c[(size_t)r * 11 + d] = a;
    }
    rt10c[(size_t)r * 11 + 10] = ct;
  }
}

// ---------------------------------------------------------------------------
// Kernel 2: spatial branch, one block per time index i. Writes out (=).
// ---------------------------------------------------------------------------
__global__ __launch_bounds__(256) void k_spatial(
    const float* __restrict__ positions, const float* __restrict__ Wg,
    const float* __restrict__ bg, const float* __restrict__ Wp,
    const float* __restrict__ bp, const float* __restrict__ tpb_all,
    const __bf16* __restrict__ g_tr, const float* __restrict__ r9c,
    float* __restrict__ out) {
  __shared__ __align__(16) __bf16 sGT[128 * 64];   // 16 KB  G_i transposed [f][k]
  __shared__ float  sS[64 * 64];                   // 16 KB  scores (f32)
  __shared__ __align__(16) __bf16 sWbf[64 * 64];   //  8 KB  weights bf16 (A)
  __shared__ __align__(16) __bf16 sWg1T[128 * 32]; //  8 KB  Wg1 transposed [f][b]
  __shared__ __align__(16) __bf16 sTheta[64 * 32]; //  4 KB  A for S0
  __shared__ __align__(16) __bf16 sPhiB[64 * 32];  //  4 KB  B' for S0 ([k][a])
  __shared__ __align__(16) __bf16 sUbf[64 * 32];   //  4 KB  U = pbar@Wp + bp
  __shared__ float  sRc[64 * 10];                  //  2.5 KB
  __shared__ float  sBg[128];                      //  512 B

  const int tid = threadIdx.x, wave = tid >> 5, lane = tid & 31;
  const int i = blockIdx.x;
  const int base = i * 64;

  // ---- async contiguous 16 KB copy: g_tr block i -> sGT (already B' layout) ----
#if USE_ASYNC_COPY
  {
    const __bf16* gsrc = g_tr + (size_t)i * 8192;
    for (int idx = tid; idx < 1024; idx += 256)
      __builtin_amdgcn_global_load_async_to_lds_b128(
          (v4i*)(gsrc + idx * 8), (v4i*)(sGT + idx * 8), 0, 0);
  }
#else
  for (int idx = tid; idx < 8192; idx += 256) sGT[idx] = g_tr[(size_t)i * 8192 + idx];
#endif

  for (int idx = tid; idx < 64 * 32; idx += 256) {
    sTheta[idx] = (__bf16)tpb_all[(size_t)(base + (idx >> 5)) * 64 + (idx & 31)];
    sPhiB[idx]  = (__bf16)tpb_all[(size_t)(base + (idx >> 5)) * 64 + 32 + (idx & 31)];
  }
  for (int idx = tid; idx < 128 * 32; idx += 256)
    sWg1T[idx] = (__bf16)Wg[(128 + (idx & 31)) * 128 + (idx >> 5)];
  for (int idx = tid; idx < 64 * 10; idx += 256)
    sRc[idx] = r9c[(size_t)base * 10 + idx];
  if (tid < 128) sBg[tid] = bg[tid];
#if USE_ASYNC_COPY
  wait_async0();
#endif
  __syncthreads();

  // S0 = Theta_i(64x32) @ PhiB_i^T(32x64): 16 tiles, one WMMA each (K=32).
  for (int t = wave; t < 16; t += 8) {
    const int mt = t >> 2, nt = t & 3;
    v16bf a = ld_frag_a(sTheta + mt * 16 * 32, 32, lane);
    v16bf b = ld_frag_bT(sPhiB + nt * 16 * 32, 32, lane);
    v8f acc = {};
    acc = wmma_bf16(a, b, acc);
    const int n = nt * 16 + (lane & 15);
#pragma unroll
    for (int v = 0; v < 8; ++v) {
      const int m = mt * 16 + v + ((lane < 16) ? 0 : 8);
      sS[m * 64 + n] = acc[v];
    }
  }
  __syncthreads();

  // scores[j][k] += positions[i,j,k,:] . R9[j] + c[j]
  for (int p = tid; p < 4096; p += 256) {
    const int j = p >> 6, k = p & 63;
    const float* pr = positions + (((size_t)i * 64 + j) * 64 + k) * 9;
    if (p + 256 < 4096)
      __builtin_prefetch(positions + (((size_t)i * 64 + ((p + 256) >> 6)) * 64 +
                                      ((p + 256) & 63)) * 9, 0, 1);
    float s = sS[p] + sRc[j * 10 + 9];
#pragma unroll
    for (int d = 0; d < 9; ++d) s += pr[d] * sRc[j * 10 + d];
    sS[p] = s;
  }
  __syncthreads();

  // Row softmax (wave32 shuffles) + pbar(9) + U = pbar@Wp + bp.
  for (int jr = 0; jr < 8; ++jr) {
    const int j = wave * 8 + jr;
    const float x0 = sS[j * 64 + lane], x1 = sS[j * 64 + 32 + lane];
    const float mx = wave_max(fmaxf(x0, x1));
    const float e0 = __expf(x0 - mx), e1 = __expf(x1 - mx);
    const float inv = 1.f / wave_sum(e0 + e1);
    const float w0 = e0 * inv, w1 = e1 * inv;
    sWbf[j * 64 + lane] = (__bf16)w0;
    sWbf[j * 64 + 32 + lane] = (__bf16)w1;
    const float* p0 = positions + (((size_t)i * 64 + j) * 64 + lane) * 9;
    const float* p1 = p0 + 32 * 9;
    float pbar[9];
#pragma unroll
    for (int d = 0; d < 9; ++d) pbar[d] = wave_sum(w0 * p0[d] + w1 * p1[d]);
    float u = bp[lane];
#pragma unroll
    for (int d = 0; d < 9; ++d) u += pbar[d] * Wp[d * 32 + lane];
    sUbf[j * 32 + lane] = (__bf16)u;
  }
  __syncthreads();

  // out[i,j,:] = W(64x64)@G(64x128) + U(64x32)@Wg1(32x128) + bg
  for (int t = wave; t < 32; t += 8) {
    const int mt = t >> 3, nt = t & 7;
    v8f acc = {};
#pragma unroll
    for (int kk = 0; kk < 2; ++kk) {
      v16bf a = ld_frag_a(sWbf + mt * 16 * 64 + kk * 32, 64, lane);
      v16bf b = ld_frag_bT(sGT + nt * 16 * 64 + kk * 32, 64, lane);
      acc = wmma_bf16(a, b, acc);
    }
    {
      v16bf a = ld_frag_a(sUbf + mt * 16 * 32, 32, lane);
      v16bf b = ld_frag_bT(sWg1T + nt * 16 * 32, 32, lane);
      acc = wmma_bf16(a, b, acc);
    }
    const int n = nt * 16 + (lane & 15);
    const float bias = sBg[n];
#pragma unroll
    for (int v = 0; v < 8; ++v) {
      const int m = mt * 16 + v + ((lane < 16) ? 0 : 8);
      out[(size_t)(i * 64 + m) * 128 + n] = acc[v] + bias;
    }
  }
}

// ---------------------------------------------------------------------------
// Kernel 3: temporal branch, one block per query time j. RMWs out (+=).
// Contractions here are batched per person (elementwise in i) -> VALU.
// ---------------------------------------------------------------------------
__global__ __launch_bounds__(256) void k_temporal(
    const float* __restrict__ tempten, const float* __restrict__ Wg,
    const float* __restrict__ bg, const float* __restrict__ Wt,
    const float* __restrict__ bt, const float* __restrict__ tpb_all,
    const __bf16* __restrict__ g_bf, const float* __restrict__ rt10c,
    float* __restrict__ out) {
  __shared__ float  sS[64 * 64];                   // 16 KB scores[i][k] -> weights
  __shared__ float  sTh[64 * 32];                  //  8 KB theta[j,i,:]
  __shared__ float  sU2[64 * 32];                  //  8 KB U2 = tbar@Wt + bt
  __shared__ __align__(16) __bf16 sWg1T[128 * 32]; //  8 KB Wg1 transposed [f][b]
  __shared__ float  sRt[64 * 11];                  //  2.8 KB

  const int tid = threadIdx.x, wave = tid >> 5, lane = tid & 31;
  const int j = blockIdx.x;
  const int base = j * 64;

  for (int idx = tid; idx < 64 * 32; idx += 256)
    sTh[idx] = tpb_all[(size_t)(base + (idx >> 5)) * 64 + (idx & 31)];
  for (int idx = tid; idx < 64 * 11; idx += 256)
    sRt[idx] = rt10c[(size_t)base * 11 + idx];
  for (int idx = tid; idx < 128 * 32; idx += 256)
    sWg1T[idx] = (__bf16)Wg[(128 + (idx & 31)) * 128 + (idx >> 5)];
  __syncthreads();

  // scores[i][k] = theta[j,i].PhiB[k,i] + temp[j,k,i,:].Rt[i] + ct[i]
  for (int p = tid; p < 4096; p += 256) {
    const int ii = p >> 6, k = p & 63;
    const float* ph = tpb_all + ((size_t)k * 64 + ii) * 64 + 32;
    float s = sRt[ii * 11 + 10];
#pragma unroll
    for (int a = 0; a < 32; ++a) s += sTh[ii * 32 + a] * ph[a];
    const float* tr = tempten + (((size_t)j * 64 + k) * 64 + ii) * 10;
#pragma unroll
    for (int d = 0; d < 10; ++d) s += tr[d] * sRt[ii * 11 + d];
    sS[ii * 64 + k] = s;
  }
  __syncthreads();

  // Softmax over k + tbar(10) + U2 = tbar@Wt + bt.
  for (int ir = 0; ir < 8; ++ir) {
    const int ii = wave * 8 + ir;
    const float x0 = sS[ii * 64 + lane], x1 = sS[ii * 64 + 32 + lane];
    const float mx = wave_max(fmaxf(x0, x1));
    const float e0 = __expf(x0 - mx), e1 = __expf(x1 - mx);
    const float inv = 1.f / wave_sum(e0 + e1);
    const float w0 = e0 * inv, w1 = e1 * inv;
    sS[ii * 64 + lane] = w0;
    sS[ii * 64 + 32 + lane] = w1;
    const float* t0 = tempten + (((size_t)j * 64 + lane) * 64 + ii) * 10;
    const float* t1 = tempten + (((size_t)j * 64 + lane + 32) * 64 + ii) * 10;
    float tbar[10];
#pragma unroll
    for (int d = 0; d < 10; ++d) tbar[d] = wave_sum(w0 * t0[d] + w1 * t1[d]);
    float u = bt[lane];
#pragma unroll
    for (int d = 0; d < 10; ++d) u += tbar[d] * Wt[d * 32 + lane];
    sU2[ii * 32 + lane] = u;
  }
  __syncthreads();

  // out[j,i,f] += sum_k w[i,k]*G[k,i,f] + sum_b U2[i,b]*Wg1T[f,b] + bg[f]
  const int ii = tid >> 2, f0 = (tid & 3) * 32;
  float acc[32];
#pragma unroll
  for (int f = 0; f < 32; ++f) acc[f] = 0.f;
  for (int k = 0; k < 64; ++k) {
    const float wv = sS[ii * 64 + k];
    const __bf16* g = g_bf + ((size_t)k * 64 + ii) * 128 + f0;
#pragma unroll
    for (int f = 0; f < 32; ++f) acc[f] += wv * (float)g[f];
  }
#pragma unroll
  for (int f = 0; f < 32; ++f) {
    float v = acc[f] + bg[f0 + f];
#pragma unroll
    for (int b = 0; b < 32; ++b)
      v += sU2[ii * 32 + b] * (float)sWg1T[(f0 + f) * 32 + b];
    out[(size_t)(j * 64 + ii) * 128 + f0 + f] += v;
  }
}

// ---------------------------------------------------------------------------
extern "C" void kernel_launch(void* const* d_in, const int* in_sizes, int n_in,
                              void* d_out, int out_size, void* d_ws, size_t ws_size,
                              hipStream_t stream) {
  (void)in_sizes; (void)n_in; (void)out_size; (void)ws_size;
  const float* batch     = (const float*)d_in[0];
  const float* positions = (const float*)d_in[1];
  const float* tempten   = (const float*)d_in[2];
  const float* Wa = (const float*)d_in[3];
  const float* ba = (const float*)d_in[4];
  const float* Wb = (const float*)d_in[5];
  const float* bb = (const float*)d_in[6];
  const float* Wg = (const float*)d_in[7];
  const float* bg = (const float*)d_in[8];
  const float* Wp = (const float*)d_in[9];
  const float* bp = (const float*)d_in[10];
  const float* Wt = (const float*)d_in[11];
  const float* bt = (const float*)d_in[12];
  float* out = (float*)d_out;

  char* ws = (char*)d_ws;
  float*  tpb_all = (float*)(ws + 0);                 // 4096*64*4 = 1 MB
  __bf16* g_bf    = (__bf16*)(ws + (1024 << 10));     // 4096*128*2 = 1 MB
  __bf16* g_tr    = (__bf16*)(ws + (2048 << 10));     // 1 MB, [i][f][k]
  float*  r9c     = (float*)(ws + (3072 << 10));      // 4096*10*4 = 160 KB
  float*  rt10c   = (float*)(ws + (3232 << 10));      // 4096*11*4 = 176 KB

  k_precompute<<<64, 256, 0, stream>>>(batch, Wa, ba, Wb, bb, Wg, Wp, bp, Wt, bt,
                                       tpb_all, g_bf, g_tr, r9c, rt10c);
  k_spatial<<<64, 256, 0, stream>>>(positions, Wg, bg, Wp, bp, tpb_all, g_tr,
                                    r9c, out);
  k_temporal<<<64, 256, 0, stream>>>(tempten, Wg, bg, Wt, bt, tpb_all, g_bf,
                                     rt10c, out);
}
